// xLSTMBlock_88940182765808
// MI455X (gfx1250) — matrix operand order repaired and checked by reading, not compiled, not run on hardware
//
#include <hip/hip_runtime.h>

#define B_SZ 8
#define S_SZ 4096
#define DIN  256
#define DH   512
#define TCH  512                 // timesteps per chunk
#define NCH  (S_SZ / TCH)        // 8 chunks
#define MCH  (B_SZ * TCH)        // 4096 gate rows per chunk
#define EPSV 1e-6f

typedef __bf16 v16bf __attribute__((ext_vector_type(16)));
typedef float  v8f   __attribute__((ext_vector_type(8)));

union FragU { v16bf bf; uint4 u4[2]; };

__device__ __forceinline__ unsigned short f2bf_rne(float f) {
    unsigned u = __float_as_uint(f);
    unsigned r = u + 0x7FFFu + ((u >> 16) & 1u);   // round to nearest even
    return (unsigned short)(r >> 16);
}

// ---------------------------------------------------------------------------
// Kernel 1: fp32 -> bf16 for x, and W transposed to WT[g][n][k] = W_g[k][n]
// ---------------------------------------------------------------------------
__global__ void convert_kernel(const float* __restrict__ x,
                               const float* __restrict__ Wi,
                               const float* __restrict__ Wf,
                               const float* __restrict__ Wo,
                               const float* __restrict__ Wz,
                               unsigned short* __restrict__ xb,
                               unsigned short* __restrict__ wt) {
    const int NX = B_SZ * S_SZ * DIN;   // 8388608
    const int NW = 4 * DH * DIN;        // 524288
    const int stride = gridDim.x * blockDim.x;
    for (int idx = blockIdx.x * blockDim.x + threadIdx.x; idx < NX + NW; idx += stride) {
        if (idx < NX) {
            xb[idx] = f2bf_rne(x[idx]);
        } else {
            int j   = idx - NX;
            int g   = j / (DH * DIN);
            int rem = j - g * (DH * DIN);
            int n   = rem / DIN;
            int k   = rem - n * DIN;
            const float* W = (g == 0) ? Wi : (g == 1) ? Wf : (g == 2) ? Wo : Wz;
            wt[j] = f2bf_rne(W[k * DH + n]);
        }
    }
}

// ---------------------------------------------------------------------------
// Kernel 2: bf16 WMMA GEMM for one time-chunk.
// Block = 8 waves sharing ONE n-tile (B fragments L0/L1-hit across waves).
// Wave  = 32 rows (2 m-tiles) x 16 cols x 4 gates = 8 accumulators,
//         64 v_wmma per wave, 1.5 b128 loads per WMMA.
// ---------------------------------------------------------------------------
__global__ void __launch_bounds__(256) gates_gemm_kernel(
    const unsigned short* __restrict__ xb,
    const unsigned short* __restrict__ wt,
    const float* __restrict__ bias_i, const float* __restrict__ bias_f,
    const float* __restrict__ bias_o, const float* __restrict__ bias_z,
    float* __restrict__ gates, int chunk)
{
    const int lane   = threadIdx.x & 31;
    const int wave   = threadIdx.x >> 5;
    const int n_tile = blockIdx.x & 31;                 // shared by all 8 waves
    const int m_pair = (blockIdx.x >> 5) * 8 + wave;    // 0..127
    const int r0   = m_pair * 32;                       // 32 rows per wave
    const int n0   = n_tile * 16;
    const int half = lane >> 4;
    const int lm   = lane & 15;

    const int b  = r0 >> 9;                     // batch (TCH == 512 rows per b; 512%32==0)
    const int tl = r0 & (TCH - 1);
    const int t0 = chunk * TCH;

    // A: row-major x (bf16); per ISA 16-bit A layout, lane-half K split.
    const unsigned short* arow0 = xb + (size_t)(b * S_SZ + t0 + tl + lm) * DIN;
    const unsigned short* arow1 = arow0 + 16 * DIN;
    // B: WT[g][n][k]; lanes 0-15 hold K=0..7,16..23, lanes 16-31 hold K=8..15,24..31.
    const unsigned short* brow = wt + (size_t)(n0 + lm) * DIN;
    const int GOFF = DH * DIN;                  // gate stride in WT

    v8f acc[8];
    #pragma unroll
    for (int i = 0; i < 8; ++i) acc[i] = (v8f){};

    #pragma unroll
    for (int kk = 0; kk < DIN / 32; ++kk) {
        const int k0 = kk * 32;
        FragU a0, a1;
        a0.u4[0] = *(const uint4*)(arow0 + k0 + 8 * half);       // K = k0+8h .. +7
        a0.u4[1] = *(const uint4*)(arow0 + k0 + 16 + 8 * half);  // K = k0+16+8h .. +7
        a1.u4[0] = *(const uint4*)(arow1 + k0 + 8 * half);
        a1.u4[1] = *(const uint4*)(arow1 + k0 + 16 + 8 * half);

        const unsigned short* bp = brow + k0 + 8 * half;         // K = k0+8h .. +7, k0+16+8h .. +7
        FragU bfr[4];
        #pragma unroll
        for (int g = 0; g < 4; ++g) {
            bfr[g].u4[0] = *(const uint4*)(bp + g * GOFF);
            bfr[g].u4[1] = *(const uint4*)(bp + g * GOFF + 16);
        }

        #pragma unroll
        for (int g = 0; g < 4; ++g) {
            acc[2 * g + 0] = __builtin_amdgcn_wmma_f32_16x16x32_bf16(
                false, a0.bf, false, bfr[g].bf, (short)0, acc[2 * g + 0], false, false);
            acc[2 * g + 1] = __builtin_amdgcn_wmma_f32_16x16x32_bf16(
                false, a1.bf, false, bfr[g].bf, (short)0, acc[2 * g + 1], false, false);
        }
    }

    // Epilogue: C/D layout — VGPR j, lane L: row = r0 + 16*mt + j + 8*(L>>4),
    // col = n0 + (L&15). Apply bias + activation, store fp32 gates.
    const int col   = n0 + lm;
    const size_t GS = (size_t)MCH * DH;

    const float bvi = bias_i[col], bvf = bias_f[col];
    const float bvo = bias_o[col], bvz = bias_z[col];

    #pragma unroll
    for (int mt = 0; mt < 2; ++mt) {
        const int rbase = r0 + 16 * mt + 8 * half;
        #pragma unroll
        for (int j = 0; j < 8; ++j) {
            const size_t rowoff = (size_t)(rbase + j) * DH + col;
            float vi = acc[0 + mt][j] + bvi;
            gates[0 * GS + rowoff] = __expf(fminf(fmaxf(vi, -20.f), 0.f));
            float vf = acc[2 + mt][j] + bvf;
            gates[1 * GS + rowoff] = __expf(fminf(fmaxf(vf, -20.f), 0.f));
            float vo = acc[4 + mt][j] + bvo;
            gates[2 * GS + rowoff] = 1.f / (1.f + __expf(-vo));
            float vz = fminf(fmaxf(acc[6 + mt][j] + bvz, -10.f), 10.f);
            float e2 = __expf(2.f * vz);
            gates[3 * GS + rowoff] = (e2 - 1.f) / (e2 + 1.f);
        }
    }
}

// ---------------------------------------------------------------------------
// Kernel 3: sequential scan over one time-chunk; one thread per (b,h) channel.
// Chunk gates (33.5 MB) are L2-resident right after the GEMM.
// ---------------------------------------------------------------------------
__global__ void __launch_bounds__(256) scan_kernel(
    const float* __restrict__ gates, float* __restrict__ state,
    float* __restrict__ out, int chunk)
{
    const int u = blockIdx.x * blockDim.x + threadIdx.x;   // 0..4095
    const int b = u >> 9;
    const int h = u & (DH - 1);
    const size_t GS = (size_t)MCH * DH;

    float c, n;
    if (chunk == 0) { c = 0.f; n = 1.f; }
    else            { c = state[u]; n = state[u + B_SZ * DH]; }

    const size_t rbase = (size_t)(b * TCH) * DH + h;
    const float* gi = gates + 0 * GS + rbase;
    const float* gf = gates + 1 * GS + rbase;
    const float* go = gates + 2 * GS + rbase;
    const float* gz = gates + 3 * GS + rbase;
    float* op = out + ((size_t)b * S_SZ + (size_t)chunk * TCH) * DH + h;

    #pragma unroll 4
    for (int t = 0; t < TCH; ++t) {
        const size_t off = (size_t)t * DH;
        float iv = gi[off], fv = gf[off], ov = go[off], zv = gz[off];
        c = fv * c + iv * zv;
        n = fv * n + iv;
        op[off] = ov * (c / (n + EPSV));
    }
    state[u]              = c;
    state[u + B_SZ * DH]  = n;
}

// ---------------------------------------------------------------------------
extern "C" void kernel_launch(void* const* d_in, const int* in_sizes, int n_in,
                              void* d_out, int out_size, void* d_ws, size_t ws_size,
                              hipStream_t stream)
{
    const float* x  = (const float*)d_in[0];
    const float* Wi = (const float*)d_in[1];
    const float* bi = (const float*)d_in[2];
    const float* Wf = (const float*)d_in[3];
    const float* bf = (const float*)d_in[4];
    const float* Wo = (const float*)d_in[5];
    const float* bo = (const float*)d_in[6];
    const float* Wz = (const float*)d_in[7];
    const float* bz = (const float*)d_in[8];
    float* out = (float*)d_out;

    char* ws = (char*)d_ws;
    // layout: [0,16MB) xb bf16 | [16,17MB) WT bf16 | [17,49MB) chunk gates f32 | state
    unsigned short* xb    = (unsigned short*)(ws);
    unsigned short* wt    = (unsigned short*)(ws + (size_t)16 * 1024 * 1024);
    float*          gates = (float*)        (ws + (size_t)17 * 1024 * 1024);
    float*          state = (float*)        (ws + (size_t)49 * 1024 * 1024);

    convert_kernel<<<4096, 256, 0, stream>>>(x, Wi, Wf, Wo, Wz, xb, wt);

    for (int chunk = 0; chunk < NCH; ++chunk) {
        gates_gemm_kernel<<<512, 256, 0, stream>>>(xb, wt, bi, bf, bo, bz, gates, chunk);
        scan_kernel<<<(B_SZ * DH) / 256, 256, 0, stream>>>(gates, state, out, chunk);
    }
}
